// OlmoLMHeadCE_72550587564595
// MI455X (gfx1250) — matrix operand order repaired and checked by reading, not compiled, not run on hardware
//
#include <hip/hip_runtime.h>
#include <stdint.h>

// Problem constants: x[2048,2048] f32, W[2048,128256] f32, labels[2048] i32
#define NROWS  2048
#define HDIM   2048
#define VDIM   128256
#define MT     128            // rows per block
#define CHUNK  256            // vocab columns per block
#define NCHUNK 501            // 501 * 256 == 128256 exactly
#define KB     32             // K per LDS stage (bf16 WMMA depth)
#define KSTEPS (HDIM / KB)    // 64
#define IGNORE_INDEX (-100)

typedef __bf16 v16bf __attribute__((ext_vector_type(16)));
typedef float  v8f   __attribute__((ext_vector_type(8)));

// fp32 -> bf16 (round to nearest even), and back
__device__ __forceinline__ unsigned short f2bf(float f) {
  unsigned u = __float_as_uint(f);
  u += 0x7FFFu + ((u >> 16) & 1u);
  return (unsigned short)(u >> 16);
}
__device__ __forceinline__ float bf2f(unsigned short b) {
  return __uint_as_float(((unsigned)b) << 16);
}
__device__ __forceinline__ unsigned pk2(unsigned short a, unsigned short b) {
  return (unsigned)a | ((unsigned)b << 16);
}
// convert one float4 -> 4 hi bf16 + 4 lo bf16
__device__ __forceinline__ void cvt4(const float4 v, unsigned short* h, unsigned short* l) {
  h[0] = f2bf(v.x); l[0] = f2bf(v.x - bf2f(h[0]));
  h[1] = f2bf(v.y); l[1] = f2bf(v.y - bf2f(h[1]));
  h[2] = f2bf(v.z); l[2] = f2bf(v.z - bf2f(h[2]));
  h[3] = f2bf(v.w); l[3] = f2bf(v.w - bf2f(h[3]));
}

union FragBF { uint4 q[2]; v16bf v; };   // 32 B = one 16x32 bf16 A/B fragment per lane

// LDS staging: uint4-typed rows (5 x 16B = 80B row stride) so fragment reads are
// true ds_load_b128 and staging writes are ds_store_b128.  K=0..31 lives in
// uint4 indices 0..3 of each row; index 4 is padding (bank-conflict stagger).
struct SharedStage {
  uint4 aHi[MT][5];       // A as [m][k]
  uint4 aLo[MT][5];
  uint4 bHi[CHUNK][5];    // B transposed: [col][k]
  uint4 bLo[CHUNK][5];
};
struct SharedRed {
  float pmax[4][MT];
  float psum[4][MT];
};
union SharedMem { SharedStage st; SharedRed rd; };

__global__ __launch_bounds__(256)
void lmhead_partial_kernel(const float* __restrict__ x,
                           const float* __restrict__ w,
                           const int*   __restrict__ labels,
                           float* __restrict__ plse,   // [NROWS][NCHUNK]
                           float* __restrict__ tgt)    // [NROWS]
{
  __shared__ SharedMem sh;

  const int tid  = threadIdx.x;
  const int lane = tid & 31;
  const int wave = tid >> 5;
  const int wm   = wave & 1;                 // 2 waves in M -> 64 rows each
  const int wn   = wave >> 1;                // 4 waves in N -> 64 cols each
  const int rowBase = blockIdx.x * MT;
  const int colBase = blockIdx.y * CHUNK;

  v8f acc[4][4];
  #pragma unroll
  for (int i = 0; i < 4; ++i)
    #pragma unroll
    for (int j = 0; j < 4; ++j)
      #pragma unroll
      for (int e = 0; e < 8; ++e) acc[i][j][e] = 0.0f;

  // ---- staging assignments ----
  // A: 2 threads per row; thread covers 16 consecutive k (one uint4-pair region)
  const int aRow = tid >> 1;
  const int aIdx = (tid & 1) * 2;            // uint4 index base (0 or 2)
  const float* xp = x + (size_t)(rowBase + aRow) * HDIM + (tid & 1) * 16;
  // B: thread covers 4 consecutive cols x 8 consecutive k  (coalesced float4 rows)
  const int bCol = (tid & 63) * 4;
  const int bKi  = tid >> 6;                 // 0..3 -> k sub-block of 8
  const float* wpB = w + (size_t)(bKi * 8) * VDIM + colBase + bCol;

  // ---- fragment addressing (ISA 16-bit A/B layouts, wave32) ----
  const int halfSel = (lane < 16) ? 0 : 1;
  const int aQ = halfSel;                    // A: q0 at idx 0/1, q1 at idx 2/3
  const int bQ = halfSel;                    // B: q0 at idx 0/1, q1 at idx 2/3
  const int lc = lane & 15;

  float4 aPre[4];                            // register prefetch: 16 floats (A)
  float4 bPre[8];                            // register prefetch: 32 floats (B)

  // load tile 0
  #pragma unroll
  for (int j = 0; j < 4; ++j) aPre[j] = *(const float4*)(xp + 4 * j);
  #pragma unroll
  for (int i = 0; i < 8; ++i) bPre[i] = *(const float4*)(wpB + (size_t)i * VDIM);

  for (int kt = 0; kt < KSTEPS; ++kt) {
    // ---- convert + store current prefetched tile into LDS ----
    {
      #pragma unroll
      for (int j = 0; j < 2; ++j) {          // A: two b128 stores per array
        unsigned short h[8], l[8];
        cvt4(aPre[2 * j],     h,     l);
        cvt4(aPre[2 * j + 1], h + 4, l + 4);
        uint4 uh; uh.x = pk2(h[0],h[1]); uh.y = pk2(h[2],h[3]);
                  uh.z = pk2(h[4],h[5]); uh.w = pk2(h[6],h[7]);
        uint4 ul; ul.x = pk2(l[0],l[1]); ul.y = pk2(l[2],l[3]);
                  ul.z = pk2(l[4],l[5]); ul.w = pk2(l[6],l[7]);
        sh.st.aHi[aRow][aIdx + j] = uh;
        sh.st.aLo[aRow][aIdx + j] = ul;
      }
      unsigned hiW[4][4], loW[4][4];         // [col][k-pair word]
      #pragma unroll
      for (int i = 0; i < 8; ++i) {          // 8 consecutive k rows
        unsigned short h[4], l[4];
        cvt4(bPre[i], h, l);
        #pragma unroll
        for (int c = 0; c < 4; ++c) {
          if ((i & 1) == 0) { hiW[c][i >> 1]  = (unsigned)h[c];
                              loW[c][i >> 1]  = (unsigned)l[c]; }
          else              { hiW[c][i >> 1] |= (unsigned)h[c] << 16;
                              loW[c][i >> 1] |= (unsigned)l[c] << 16; }
        }
      }
      #pragma unroll
      for (int c = 0; c < 4; ++c) {          // one b128 store per col per array
        uint4 uh; uh.x = hiW[c][0]; uh.y = hiW[c][1]; uh.z = hiW[c][2]; uh.w = hiW[c][3];
        uint4 ul; ul.x = loW[c][0]; ul.y = loW[c][1]; ul.z = loW[c][2]; ul.w = loW[c][3];
        sh.st.bHi[bCol + c][bKi] = uh;
        sh.st.bLo[bCol + c][bKi] = ul;
      }
    }
    __syncthreads();                          // tile kt visible to all waves

    // ---- issue global prefetch for tile kt+1 (hides latency under WMMA) ----
    if (kt + 1 < KSTEPS) {
      const int k0n = (kt + 1) * KB;
      #pragma unroll
      for (int j = 0; j < 4; ++j) aPre[j] = *(const float4*)(xp + k0n + 4 * j);
      const float* p = wpB + (size_t)k0n * VDIM;
      #pragma unroll
      for (int i = 0; i < 8; ++i) bPre[i] = *(const float4*)(p + (size_t)i * VDIM);
    }

    // ---- build fragments (b128 LDS reads) and run 48 WMMAs ----
    FragBF aHiF[4], aLoF[4];
    #pragma unroll
    for (int ms = 0; ms < 4; ++ms) {
      const int m = wm * 64 + ms * 16 + lc;
      aHiF[ms].q[0] = sh.st.aHi[m][aQ];
      aHiF[ms].q[1] = sh.st.aHi[m][aQ + 2];
      aLoF[ms].q[0] = sh.st.aLo[m][aQ];
      aLoF[ms].q[1] = sh.st.aLo[m][aQ + 2];
    }
    #pragma unroll
    for (int ns = 0; ns < 4; ++ns) {
      const int c = wn * 64 + ns * 16 + lc;
      FragBF bHiF, bLoF;
      bHiF.q[0] = sh.st.bHi[c][bQ];
      bHiF.q[1] = sh.st.bHi[c][bQ + 2];
      bLoF.q[0] = sh.st.bLo[c][bQ];
      bLoF.q[1] = sh.st.bLo[c][bQ + 2];
      #pragma unroll
      for (int ms = 0; ms < 4; ++ms) {
        // fp32-accurate product via bf16 split: Ahi*Bhi + Ahi*Blo + Alo*Bhi
        acc[ms][ns] = __builtin_amdgcn_wmma_f32_16x16x32_bf16(
            false, aHiF[ms].v, false, bHiF.v, (short)0, acc[ms][ns], false, false);
        acc[ms][ns] = __builtin_amdgcn_wmma_f32_16x16x32_bf16(
            false, aHiF[ms].v, false, bLoF.v, (short)0, acc[ms][ns], false, false);
        acc[ms][ns] = __builtin_amdgcn_wmma_f32_16x16x32_bf16(
            false, aLoF[ms].v, false, bHiF.v, (short)0, acc[ms][ns], false, false);
      }
    }
    __syncthreads();                          // all waves done reading tile kt
  }

  // ---- per-wave online logsumexp over this wave's 64 columns + target pick ----
  // C layout: VGPR r of tile (ms,ns): row = ms*16 + (lane<16 ? r : r+8), col = ns*16 + lane%16
  #pragma unroll
  for (int ms = 0; ms < 4; ++ms) {
    #pragma unroll
    for (int r = 0; r < 8; ++r) {
      const int rowLocal = wm * 64 + ms * 16 + (halfSel ? r + 8 : r);
      float mx = acc[ms][0][r];
      #pragma unroll
      for (int ns = 1; ns < 4; ++ns) mx = fmaxf(mx, acc[ms][ns][r]);
      #pragma unroll
      for (int off = 1; off < 16; off <<= 1) mx = fmaxf(mx, __shfl_xor(mx, off, 32));

      const int lab = labels[rowBase + rowLocal];
      const int rel = lab - (colBase + wn * 64);
      float s = 0.0f;
      #pragma unroll
      for (int ns = 0; ns < 4; ++ns) {
        s += __expf(acc[ms][ns][r] - mx);
        if (rel == ns * 16 + lc)              // exactly one lane in the grid matches
          tgt[rowBase + rowLocal] = acc[ms][ns][r];
      }
      #pragma unroll
      for (int off = 1; off < 16; off <<= 1) s += __shfl_xor(s, off, 32);

      if (lc == 0) { sh.rd.pmax[wn][rowLocal] = mx; sh.rd.psum[wn][rowLocal] = s; }
    }
  }
  __syncthreads();

  // ---- combine the 4 N-waves -> per-chunk partial lse for 128 rows ----
  if (tid < MT) {
    float m0 = sh.rd.pmax[0][tid], m1 = sh.rd.pmax[1][tid];
    float m2 = sh.rd.pmax[2][tid], m3 = sh.rd.pmax[3][tid];
    float M = fmaxf(fmaxf(m0, m1), fmaxf(m2, m3));
    float S = sh.rd.psum[0][tid] * __expf(m0 - M)
            + sh.rd.psum[1][tid] * __expf(m1 - M)
            + sh.rd.psum[2][tid] * __expf(m2 - M)
            + sh.rd.psum[3][tid] * __expf(m3 - M);
    plse[(size_t)(rowBase + tid) * NCHUNK + blockIdx.y] = M + __logf(S);
  }
}

__global__ __launch_bounds__(256)
void lmhead_final_kernel(const float* __restrict__ plse,
                         const float* __restrict__ tgt,
                         const int*   __restrict__ labels,
                         float* __restrict__ out)
{
  const int lane = threadIdx.x & 31;
  const int wave = threadIdx.x >> 5;
  const int row  = blockIdx.x * 8 + wave;
  const float* p = plse + (size_t)row * NCHUNK;

  float mx = -3.4e38f;
  for (int i = lane; i < NCHUNK; i += 32) mx = fmaxf(mx, p[i]);
  #pragma unroll
  for (int off = 1; off < 32; off <<= 1) mx = fmaxf(mx, __shfl_xor(mx, off, 32));

  float s = 0.0f;
  for (int i = lane; i < NCHUNK; i += 32) s += __expf(p[i] - mx);
  #pragma unroll
  for (int off = 1; off < 32; off <<= 1) s += __shfl_xor(s, off, 32);

  if (lane == 0) {
    float lse = mx + __logf(s);
    int lab = labels[row];
    out[row]         = (lab != IGNORE_INDEX) ? (lse - tgt[row]) : 0.0f;
    out[NROWS + row] = lse;
  }
}

extern "C" void kernel_launch(void* const* d_in, const int* in_sizes, int n_in,
                              void* d_out, int out_size, void* d_ws, size_t ws_size,
                              hipStream_t stream) {
  const float* x      = (const float*)d_in[0];
  const float* w      = (const float*)d_in[1];
  const int*   labels = (const int*)d_in[2];
  float* out  = (float*)d_out;
  float* plse = (float*)d_ws;                       // NROWS*NCHUNK floats (~4.1 MB)
  float* tgt  = plse + (size_t)NROWS * NCHUNK;      // NROWS floats

  dim3 grid1(NROWS / MT, NCHUNK);                   // x-fastest: 16 blocks sharing one
  lmhead_partial_kernel<<<grid1, 256, 0, stream>>>( //   weight chunk -> 192MB L2 reuse
      x, w, labels, plse, tgt);
  lmhead_final_kernel<<<NROWS / 8, 256, 0, stream>>>(plse, tgt, labels, out);
}
